// MemoryAsContextTransformer_17274358464742
// MI455X (gfx1250) — matrix operand order repaired and checked by reading, not compiled, not run on hardware
//
#include <hip/hip_runtime.h>
#include <hip/hip_bf16.h>

// ---------------------------------------------------------------------------
// MI455X (gfx1250) MAC-transformer forward.
// bf16 WMMA (v_wmma_f32_16x16x32_bf16) for all three matmul stages,
// flash-attention online softmax, LDS-staged tiles (async global->LDS for the
// non-transposed panels), precomputed rotary cos/sin tables (L2-resident).
// ---------------------------------------------------------------------------

typedef __attribute__((ext_vector_type(16))) __bf16 v16bf;
typedef __attribute__((ext_vector_type(8)))  float  v8f;

struct Frag { union { v16bf v; unsigned int u[8]; }; };

#define N_HEADS   16
#define DH        64
#define SEGLEN    512
#define PMEM      16
#define KVROWS    544            // 16 pm + 512 tokens + 16 pad (masked)
#define NSEG      16             // b(2) * segments-per-batch(8)
#define DMODEL    1024
#define NROWS     8192           // b * n
#define NTOK      4096           // tokens per batch (rotary positions)
#define NEG_INF  (-1.0e30f)

// --- fragment index helpers (wave32, 16-bit WMMA layouts per CDNA5 ISA) -----
// A 16x32: lane L holds row m=L%16; VGPR v, half h -> k = aBase(v) + 8*(L/16) + h
__device__ __forceinline__ int a_kbase(int v, int lane) {
    int base = (v < 4) ? (v * 2) : (16 + (v - 4) * 2);
    return base + ((lane >> 4) << 3);
}
// B 32x16: lane L holds col n=L%16; VGPR v, half h -> k = aBase(v) + 8*(L/16) + h
__device__ __forceinline__ int b_kbase(int v, int lane) {
    int base = (v < 4) ? (v * 2) : (16 + (v - 4) * 2);
    return base + ((lane >> 4) << 3);
}

__device__ __forceinline__ float rmax16(float x) {
    for (int m = 1; m < 16; m <<= 1) x = fmaxf(x, __shfl_xor(x, m, 32));
    return x;
}
__device__ __forceinline__ float rsum16(float x) {
    for (int m = 1; m < 16; m <<= 1) x += __shfl_xor(x, m, 32);
    return x;
}

// --- gfx1250 async global->LDS (ASYNCcnt-tracked, bypasses VGPRs) -----------
// LDS destination = low 32 bits of the generic shared pointer (flat-LDS
// aperture truncation); global address in a VGPR pair, GV addressing mode.
__device__ __forceinline__ void async_load_b128(const void* gptr, void* lptr) {
    unsigned lofs = (unsigned)(uintptr_t)lptr;
    unsigned long long gaddr = (unsigned long long)(uintptr_t)gptr;
    asm volatile("global_load_async_to_lds_b128 %0, %1, off"
                 :: "v"(lofs), "v"(gaddr)
                 : "memory");
}
__device__ __forceinline__ void wait_async0() {
    asm volatile("s_wait_asynccnt 0" ::: "memory");
}

// ---------------------------------------------------------------------------
// Stage 0a: elementwise fp32 -> bf16 cast (weights)
// ---------------------------------------------------------------------------
__global__ void cast_bf16_kernel(const float* __restrict__ in,
                                 __bf16* __restrict__ out, int n) {
    int i = blockIdx.x * 256 + threadIdx.x;
    if (i < n) out[i] = (__bf16)in[i];
}

// Stage 0b: rotary cos/sin tables [4096][64] (2 MB total, L2-resident).
__global__ void rope_table_kernel(float* __restrict__ ct,
                                  float* __restrict__ st) {
    int idx = blockIdx.x * 256 + threadIdx.x;        // 4096*64
    if (idx >= NTOK * DH) return;
    int d = idx & 63, tpos = idx >> 6;
    float fr  = __powf(10000.0f, -(float)(d & ~1) / 64.0f);
    float ang = (float)tpos * fr;
    ct[idx] = __cosf(ang);
    st[idx] = __sinf(ang);
}

// Stage 0c: persistent-memory K/V broadcast into every segment's KV buffer.
__global__ void pm_fill_kernel(const float* __restrict__ pm,
                               __bf16* __restrict__ kseg,
                               __bf16* __restrict__ vseg) {
    int idx = blockIdx.x * 256 + threadIdx.x;        // 2*16*16*64 = 32768
    if (idx >= 2 * N_HEADS * PMEM * DH) return;
    int d = idx & 63;
    int r = (idx >> 6) & 15;
    int h = (idx >> 10) & 15;
    int t = idx >> 14;
    __bf16 val = (__bf16)pm[idx];
    __bf16* dst = t ? vseg : kseg;
    for (int s = 0; s < NSEG; ++s)
        dst[(((size_t)(s * N_HEADS + h)) * KVROWS + r) * DH + d] = val;
}

// ---------------------------------------------------------------------------
// Stage 1: RMSNorm (eps = fp32 eps) * gamma, cast to bf16
// ---------------------------------------------------------------------------
__global__ __launch_bounds__(256) void rmsnorm_kernel(const float* __restrict__ seq,
                                                      const float* __restrict__ g,
                                                      __bf16* __restrict__ xb) {
    int row = blockIdx.x;                            // 8192
    const float* x = seq + (size_t)row * DMODEL;
    float ss = 0.f;
    for (int c = threadIdx.x; c < DMODEL; c += 256) { float v = x[c]; ss += v * v; }
    for (int m = 1; m < 32; m <<= 1) ss += __shfl_xor(ss, m, 32);
    __shared__ float red[8];
    if ((threadIdx.x & 31) == 0) red[threadIdx.x >> 5] = ss;
    __syncthreads();
    float tot = 0.f;
    for (int i = 0; i < 8; ++i) tot += red[i];
    float inv = rsqrtf(tot / (float)DMODEL + 1.1920929e-7f);
    for (int c = threadIdx.x; c < DMODEL; c += 256)
        xb[(size_t)row * DMODEL + c] = (__bf16)(x[c] * inv * g[c]);
}

// ---------------------------------------------------------------------------
// Stage 2: QKV GEMM (8192x1024 @ 1024x3072, bf16 WMMA) fused with rotary
// (table lookup) and scatter into segmented head-major q/k/v buffers.
// Block: 128 thr (4 waves), 64x64 tile; wave tile 32x32 (2x2 WMMA accums).
// ---------------------------------------------------------------------------
__global__ __launch_bounds__(128) void gemm_qkv_rope_kernel(
        const __bf16* __restrict__ xb, const __bf16* __restrict__ wb,
        const float* __restrict__ ct, const float* __restrict__ st,
        __bf16* __restrict__ qseg, __bf16* __restrict__ kseg,
        __bf16* __restrict__ vseg) {
    const int N = 3 * DMODEL, K = DMODEL;
    int bn = blockIdx.x * 64, bm = blockIdx.y * 64;
    int wave = threadIdx.x >> 5, lane = threadIdx.x & 31;
    int wm = (wave >> 1) * 32, wn = (wave & 1) * 32;
    int ln = lane & 15;

    __shared__ __bf16 As[64][32];     // A row-major
    __shared__ __bf16 BsT[64][32];    // B transposed: BsT[n][k]

    v8f acc[2][2] = {};

    for (int kk = 0; kk < K; kk += 32) {
        { // stage A via async global->LDS (ASYNCcnt), B transposed in-register
            int r = threadIdx.x >> 1, half = (threadIdx.x & 1) * 16;
            const __bf16* src = xb + (size_t)(bm + r) * K + kk + half;
            async_load_b128(src,     &As[r][half]);
            async_load_b128(src + 8, &As[r][half + 8]);
            int br = threadIdx.x >> 2, bq = (threadIdx.x & 3) * 16;
            const __bf16* bsrc = wb + (size_t)(kk + br) * N + bn + bq;
            for (int e = 0; e < 16; ++e) BsT[bq + e][br] = bsrc[e];
        }
        wait_async0();
        __syncthreads();

        Frag a[2], b[2];
        for (int t = 0; t < 2; ++t)
            for (int v = 0; v < 8; ++v) {
                a[t].u[v] = *(const unsigned int*)&As[wm + t * 16 + ln][a_kbase(v, lane)];
                b[t].u[v] = *(const unsigned int*)&BsT[wn + t * 16 + ln][b_kbase(v, lane)];
            }
        for (int mt = 0; mt < 2; ++mt)
            for (int nt = 0; nt < 2; ++nt)
                acc[mt][nt] = __builtin_amdgcn_wmma_f32_16x16x32_bf16(
                    false, a[mt].v, false, b[nt].v, (short)0, acc[mt][nt], false, false);
        __syncthreads();
    }

    // Epilogue: rotary (q,k) via table + scatter-store bf16 into segments
    for (int mt = 0; mt < 2; ++mt)
        for (int nt = 0; nt < 2; ++nt) {
            int cb  = bn + wn + nt * 16;            // tile never straddles q/k/v or head
            int sct = cb / DMODEL;                  // 0=q 1=k 2=v
            int hh  = (cb % DMODEL) / DH;
            int d   = (cb % DH) + ln;
            for (int r = 0; r < 8; ++r) {
                int mrow = bm + wm + mt * 16 + r + ((lane >> 4) << 3);
                int bidx = mrow >> 12;              // /4096
                int tpos = mrow & 4095;
                float val = acc[mt][nt][r];
                float partner = __shfl_xor(val, 1, 32);
                if (sct < 2) {
                    float cs = ct[tpos * DH + d];
                    float sn = st[tpos * DH + d];
                    val = (d & 1) ? (val * cs + partner * sn)
                                  : (val * cs - partner * sn);
                }
                int sseg = bidx * 8 + (tpos >> 9);
                int i    = tpos & 511;
                size_t hb = (size_t)(sseg * N_HEADS + hh);
                if (sct == 0)
                    qseg[(hb * SEGLEN + i) * DH + d] = (__bf16)val;
                else if (sct == 1)
                    kseg[(hb * KVROWS + PMEM + i) * DH + d] = (__bf16)val;
                else
                    vseg[(hb * KVROWS + PMEM + i) * DH + d] = (__bf16)val;
            }
        }
}

// ---------------------------------------------------------------------------
// Stage 3: flash attention per (segment, head, 64-row block).
// 128 thr = 4 waves; wave w owns the 16-row tile [mb*64+w*16, +16).
// ---------------------------------------------------------------------------
__global__ __launch_bounds__(128) void attn_kernel(
        const __bf16* __restrict__ qs, const __bf16* __restrict__ ks,
        const __bf16* __restrict__ vs, __bf16* __restrict__ ob) {
    int mb = blockIdx.x, h = blockIdx.y, s = blockIdx.z;
    int wave = threadIdx.x >> 5, lane = threadIdx.x & 31;
    int ln = lane & 15, kh8 = (lane >> 4) << 3;
    int i0 = mb * 64 + wave * 16;

    __shared__ __bf16 Kt[32][64];        // K panel row-major [j][d]
    __shared__ __bf16 VT[64][32];        // V panel transposed [d][j]
    __shared__ __bf16 Pscr[4][16][32];   // per-wave P relayout scratch

    size_t hb = (size_t)(s * N_HEADS + h);
    const __bf16* qbase = qs + hb * SEGLEN * DH;
    const __bf16* kbase = ks + hb * KVROWS * DH;
    const __bf16* vbase = vs + hb * KVROWS * DH;

    // Q fragments (16x64 -> two 16x32 A-frags), gathered straight from global
    Frag aq[2];
    for (int f = 0; f < 2; ++f)
        for (int v = 0; v < 8; ++v) {
            int k = a_kbase(v, lane) + f * 32;
            aq[f].u[v] = *(const unsigned int*)&qbase[(size_t)(i0 + ln) * DH + k];
        }

    float mrow[8], lrow[8];
    v8f oacc[4] = {};
    for (int r = 0; r < 8; ++r) { mrow[r] = NEG_INF; lrow[r] = 0.f; }

    int jmax = PMEM + mb * 64 + 64; if (jmax > PMEM + SEGLEN) jmax = PMEM + SEGLEN;
    int jsteps = (jmax + 31) / 32;

    for (int js = 0; js < jsteps; ++js) {
        int jb = js * 32;
        { // K panel via async global->LDS; V transposed in-register
            int jr = threadIdx.x >> 2, dq = (threadIdx.x & 3) * 16;
            const __bf16* src = kbase + (size_t)(jb + jr) * DH + dq;
            async_load_b128(src,     &Kt[jr][dq]);
            async_load_b128(src + 8, &Kt[jr][dq + 8]);
            const __bf16* vsrc = vbase + (size_t)(jb + jr) * DH + dq;
            for (int e = 0; e < 16; ++e) VT[dq + e][jr] = vsrc[e];
        }
        wait_async0();
        __syncthreads();

        // S = (Q K^T) * dh^-0.5 for two 16-col tiles
        v8f s0 = {}, s1 = {};
        {
            Frag b0a, b0b, b1a, b1b;
            for (int v = 0; v < 8; ++v) {
                int dk = b_kbase(v, lane);
                b0a.u[v] = *(const unsigned int*)&Kt[ln][dk];
                b0b.u[v] = *(const unsigned int*)&Kt[ln][32 + dk];
                b1a.u[v] = *(const unsigned int*)&Kt[16 + ln][dk];
                b1b.u[v] = *(const unsigned int*)&Kt[16 + ln][32 + dk];
            }
            s0 = __builtin_amdgcn_wmma_f32_16x16x32_bf16(false, aq[0].v, false, b0a.v, (short)0, s0, false, false);
            s0 = __builtin_amdgcn_wmma_f32_16x16x32_bf16(false, aq[1].v, false, b0b.v, (short)0, s0, false, false);
            s1 = __builtin_amdgcn_wmma_f32_16x16x32_bf16(false, aq[0].v, false, b1a.v, (short)0, s1, false, false);
            s1 = __builtin_amdgcn_wmma_f32_16x16x32_bf16(false, aq[1].v, false, b1b.v, (short)0, s1, false, false);
        }

        // online softmax update per row (row = r + 8*(lane/16))
        for (int r = 0; r < 8; ++r) {
            int ig = i0 + r + kh8;
            int j0 = jb + ln, j1 = jb + 16 + ln;
            float v0 = s0[r] * 0.125f;
            float v1 = s1[r] * 0.125f;
            bool vis0 = (j0 < PMEM) || ((j0 - PMEM) <= ig && j0 < PMEM + SEGLEN);
            bool vis1 = (j1 < PMEM) || ((j1 - PMEM) <= ig && j1 < PMEM + SEGLEN);
            v0 = vis0 ? v0 : NEG_INF;
            v1 = vis1 ? v1 : NEG_INF;
            float tm = rmax16(fmaxf(v0, v1));
            float mn = fmaxf(mrow[r], tm);
            float sc = __expf(mrow[r] - mn);
            float e0 = __expf(v0 - mn), e1 = __expf(v1 - mn);
            float ts = rsum16(e0 + e1);
            lrow[r] = lrow[r] * sc + ts;
            mrow[r] = mn;
            for (int nb = 0; nb < 4; ++nb) oacc[nb][r] = oacc[nb][r] * sc;
            int mm = r + kh8;
            Pscr[wave][mm][ln]      = (__bf16)e0;
            Pscr[wave][mm][16 + ln] = (__bf16)e1;
        }
        __builtin_amdgcn_wave_barrier();   // keep compiler from hoisting loads

        // O += P @ V  (P relayout via per-wave LDS; LDS is in-order per wave)
        Frag ap;
        for (int v = 0; v < 8; ++v)
            ap.u[v] = *(const unsigned int*)&Pscr[wave][ln][a_kbase(v, lane)];
        for (int nb = 0; nb < 4; ++nb) {
            Frag bv;
            for (int v = 0; v < 8; ++v)
                bv.u[v] = *(const unsigned int*)&VT[nb * 16 + ln][b_kbase(v, lane)];
            oacc[nb] = __builtin_amdgcn_wmma_f32_16x16x32_bf16(
                false, ap.v, false, bv.v, (short)0, oacc[nb], false, false);
        }
        __syncthreads();
    }

    // normalize + scatter to [8192,1024] bf16 for the output projection
    int bidx = s >> 3, w = s & 7;
    for (int nb = 0; nb < 4; ++nb)
        for (int r = 0; r < 8; ++r) {
            int ig = i0 + r + kh8;
            float o = oacc[nb][r] / lrow[r];
            size_t tok = (size_t)bidx * NTOK + w * SEGLEN + ig;
            int col = h * DH + nb * 16 + ln;
            ob[tok * DMODEL + col] = (__bf16)o;
        }
}

// ---------------------------------------------------------------------------
// Stage 4: output GEMM (8192x1024 @ 1024x1024, bf16 WMMA, fp32 out)
// ---------------------------------------------------------------------------
__global__ __launch_bounds__(128) void gemm_out_kernel(
        const __bf16* __restrict__ ab, const __bf16* __restrict__ wb,
        float* __restrict__ out) {
    const int N = DMODEL, K = DMODEL;
    int bn = blockIdx.x * 64, bm = blockIdx.y * 64;
    int wave = threadIdx.x >> 5, lane = threadIdx.x & 31;
    int wm = (wave >> 1) * 32, wn = (wave & 1) * 32;
    int ln = lane & 15;

    __shared__ __bf16 As[64][32];
    __shared__ __bf16 BsT[64][32];

    v8f acc[2][2] = {};

    for (int kk = 0; kk < K; kk += 32) {
        {
            int r = threadIdx.x >> 1, half = (threadIdx.x & 1) * 16;
            const __bf16* src = ab + (size_t)(bm + r) * K + kk + half;
            async_load_b128(src,     &As[r][half]);
            async_load_b128(src + 8, &As[r][half + 8]);
            int br = threadIdx.x >> 2, bq = (threadIdx.x & 3) * 16;
            const __bf16* bsrc = wb + (size_t)(kk + br) * N + bn + bq;
            for (int e = 0; e < 16; ++e) BsT[bq + e][br] = bsrc[e];
        }
        wait_async0();
        __syncthreads();
        Frag a[2], b[2];
        for (int t = 0; t < 2; ++t)
            for (int v = 0; v < 8; ++v) {
                a[t].u[v] = *(const unsigned int*)&As[wm + t * 16 + ln][a_kbase(v, lane)];
                b[t].u[v] = *(const unsigned int*)&BsT[wn + t * 16 + ln][b_kbase(v, lane)];
            }
        for (int mt = 0; mt < 2; ++mt)
            for (int nt = 0; nt < 2; ++nt)
                acc[mt][nt] = __builtin_amdgcn_wmma_f32_16x16x32_bf16(
                    false, a[mt].v, false, b[nt].v, (short)0, acc[mt][nt], false, false);
        __syncthreads();
    }
    for (int mt = 0; mt < 2; ++mt)
        for (int nt = 0; nt < 2; ++nt) {
            int col = bn + wn + nt * 16 + ln;
            for (int r = 0; r < 8; ++r) {
                int mrow = bm + wm + mt * 16 + r + ((lane >> 4) << 3);
                out[(size_t)mrow * N + col] = acc[mt][nt][r];
            }
        }
}

// ---------------------------------------------------------------------------
extern "C" void kernel_launch(void* const* d_in, const int* in_sizes, int n_in,
                              void* d_out, int out_size, void* d_ws, size_t ws_size,
                              hipStream_t stream) {
    const float* seq   = (const float*)d_in[0];   // [2,4096,1024]
    const float* gnorm = (const float*)d_in[1];   // [1024]
    const float* wqkv  = (const float*)d_in[2];   // [1024,3072]
    const float* wout  = (const float*)d_in[3];   // [1024,1024]
    const float* pm    = (const float*)d_in[4];   // [2,16,16,64]
    float* out = (float*)d_out;

    char* ws = (char*)d_ws;
    size_t off = 0;
    auto carve = [&](size_t bytes) -> char* {
        char* p = ws + off;
        off = (off + bytes + 255) & ~(size_t)255;
        return p;
    };
    __bf16* xb    = (__bf16*)carve((size_t)NROWS * DMODEL * 2);
    __bf16* wqkvb = (__bf16*)carve((size_t)DMODEL * 3 * DMODEL * 2);
    __bf16* woutb = (__bf16*)carve((size_t)DMODEL * DMODEL * 2);
    __bf16* qseg  = (__bf16*)carve((size_t)NSEG * N_HEADS * SEGLEN * DH * 2);
    __bf16* kseg  = (__bf16*)carve((size_t)NSEG * N_HEADS * KVROWS * DH * 2);
    __bf16* vseg  = (__bf16*)carve((size_t)NSEG * N_HEADS * KVROWS * DH * 2);
    __bf16* ob    = (__bf16*)carve((size_t)NROWS * DMODEL * 2);
    float*  ct    = (float*) carve((size_t)NTOK * DH * 4);
    float*  st    = (float*) carve((size_t)NTOK * DH * 4);

    // Stage 0: weight casts + rope tables + pm broadcast
    cast_bf16_kernel<<<(DMODEL * 3 * DMODEL + 255) / 256, 256, 0, stream>>>(wqkv, wqkvb, DMODEL * 3 * DMODEL);
    cast_bf16_kernel<<<(DMODEL * DMODEL + 255) / 256, 256, 0, stream>>>(wout, woutb, DMODEL * DMODEL);
    rope_table_kernel<<<(NTOK * DH + 255) / 256, 256, 0, stream>>>(ct, st);
    pm_fill_kernel<<<(2 * N_HEADS * PMEM * DH + 255) / 256, 256, 0, stream>>>(pm, kseg, vseg);

    // Stage 1: RMSNorm -> bf16
    rmsnorm_kernel<<<NROWS, 256, 0, stream>>>(seq, gnorm, xb);

    // Stage 2: QKV GEMM + rotary + segment scatter
    gemm_qkv_rope_kernel<<<dim3(3 * DMODEL / 64, NROWS / 64), 128, 0, stream>>>(
        xb, wqkvb, ct, st, qseg, kseg, vseg);

    // Stage 3: flash attention
    attn_kernel<<<dim3(SEGLEN / 64, N_HEADS, NSEG), 128, 0, stream>>>(qseg, kseg, vseg, ob);

    // Stage 4: output projection (fp32 out)
    gemm_out_kernel<<<dim3(DMODEL / 64, NROWS / 64), 128, 0, stream>>>(ob, woutb, out);
}
